// NeighborGatedAttention_7834020348392
// MI455X (gfx1250) — matrix operand order repaired and checked by reading, not compiled, not run on hardware
//
#include <hip/hip_runtime.h>

// ---------------- problem constants (match reference) ----------------
#define LNUM   2
#define NNEI   128
#define EMB    128
#define HID    128
#define SHIFT  20.0f
#define SCALING 0.08838834764831845f   // 128^-0.5
#define LN_EPS 1e-5f

#define LDH 136   // padded f16 row stride (halves): kills bank conflicts on row-strided reads
#define LDC 128   // f32 scratch row stride (floats)
#define WLD 136   // staged-weight (transposed) row stride

typedef _Float16 half_t;
typedef __attribute__((ext_vector_type(16))) _Float16 v16h;
typedef __attribute__((ext_vector_type(8)))  float    v8f;

// ---------------- static LDS layout (~301 KB of the 320 KB/WGP) ----------------
struct __align__(16) Smem {
  float  resid[NNEI * EMB];      //  64 KB   residual (f32)
  float  buf[NNEI * LDC];        //  64 KB   f32 GEMM scratch (qkv chunk / attn / o / out)
  half_t xh[NNEI * LDH];         //  34 KB   layer input f16 (reused as o-f16)
  half_t qh[NNEI * LDH];         //  34 KB   q (reused as attn-weight matrix)
  half_t kh[NNEI * LDH];         //  34 KB   k (row-major == B^T layout for q@k^T)
  half_t vhT[HID * LDH];         //  34 KB   v stored transposed (h-major)
  half_t wbufT[EMB * WLD];       //  34 KB   staged weight chunk, transposed
  float  s_sw[NNEI];
  float  s_r[NNEI * 4];
  float  s_mask[NNEI];
};

// ---------------- wave reductions (wave32) ----------------
__device__ inline float wave_sum(float v) {
#pragma unroll
  for (int o = 16; o >= 1; o >>= 1) v += __shfl_xor(v, o, 32);
  return v;
}
__device__ inline float wave_max(float v) {
#pragma unroll
  for (int o = 16; o >= 1; o >>= 1) v = fmaxf(v, __shfl_xor(v, o, 32));
  return v;
}

// ======== WMMA fragment macros: DIRECT indexing on __shared__ arrays ========
// (no pointer-typed intermediates -> guaranteed addrspace(3) DS addressing)

// A: 16x32 f16 (ISA 7.12.2): lane groups g=0/1 both M=0..15; half j -> K=((j<8)?j:j+8)+g*8
#define LOAD_A_FRAG(dst, Aarr, lda, k0)                                              \
  {                                                                                  \
    const int baseA_ = (wave * 16 + (lane & 15)) * (lda) + (k0) + ((lane >> 4) & 1) * 8; \
    _Pragma("unroll") for (int j_ = 0; j_ < 16; ++j_)                                \
        dst[j_] = Aarr[baseA_ + ((j_ < 8) ? j_ : (j_ + 8))];                         \
  }

// B: 32x16 f16 stored column-major (Barr[n*ldb + k] == B[k][n]); half j -> K=((j<8)?j:j+8)+g*8 (matches A)
#define LOAD_B_FRAG(dst, Barr, ldb, k0, n0)                                          \
  {                                                                                  \
    const int baseB_ = ((n0) + (lane & 15)) * (ldb) + (k0) + ((lane >> 4) & 1) * 8; \
    _Pragma("unroll") for (int j_ = 0; j_ < 16; ++j_) dst[j_] = Barr[baseB_ + ((j_ < 8) ? j_ : (j_ + 8))];   \
  }

// D/C: 16x16 f32, lane n = lane&15, VGPR i -> row m0 + i + 8*(lane>>4)
#define STORE_D_FRAG(Carr, ldc, n0, accv)                                            \
  {                                                                                  \
    const int nD_ = (n0) + (lane & 15);                                              \
    const int mD_ = wave * 16 + ((lane >> 4) & 1) * 8;                               \
    _Pragma("unroll") for (int i_ = 0; i_ < 8; ++i_)                                 \
        Carr[(mD_ + i_) * (ldc) + nD_] = accv[i_];                                   \
  }

#define WMMA_F16(acc, a, b)                                                          \
  acc = __builtin_amdgcn_wmma_f32_16x16x32_f16(false, a, false, b, (short)0, acc, false, false)

// 128x128x128 GEMM, wave-tiled: 16 rows/wave, two N-tiles in flight (independent
// accumulators interleave WMMAs -> scheduler fills WMMA hazard slots, no v_nops).
#define GEMM128(Aarr, lda, Barr, ldb, HASB, biasptr)                                 \
  {                                                                                  \
    v16h a0_, a1_, a2_, a3_;                                                         \
    LOAD_A_FRAG(a0_, Aarr, lda, 0)                                                   \
    LOAD_A_FRAG(a1_, Aarr, lda, 32)                                                  \
    LOAD_A_FRAG(a2_, Aarr, lda, 64)                                                  \
    LOAD_A_FRAG(a3_, Aarr, lda, 96)                                                  \
    _Pragma("unroll") for (int tp_ = 0; tp_ < 4; ++tp_) {                            \
      const int n0_ = tp_ * 32, n1_ = n0_ + 16;                                      \
      v8f acc0_ = {0.f, 0.f, 0.f, 0.f, 0.f, 0.f, 0.f, 0.f};                          \
      v8f acc1_ = {0.f, 0.f, 0.f, 0.f, 0.f, 0.f, 0.f, 0.f};                          \
      v16h b00_, b01_, b10_, b11_;                                                   \
      LOAD_B_FRAG(b00_, Barr, ldb, 0, n0_)                                           \
      LOAD_B_FRAG(b01_, Barr, ldb, 0, n1_)                                           \
      LOAD_B_FRAG(b10_, Barr, ldb, 32, n0_)                                          \
      LOAD_B_FRAG(b11_, Barr, ldb, 32, n1_)                                          \
      WMMA_F16(acc0_, a0_, b00_);                                                    \
      WMMA_F16(acc1_, a0_, b01_);                                                    \
      WMMA_F16(acc0_, a1_, b10_);                                                    \
      WMMA_F16(acc1_, a1_, b11_);                                                    \
      LOAD_B_FRAG(b00_, Barr, ldb, 64, n0_)                                          \
      LOAD_B_FRAG(b01_, Barr, ldb, 64, n1_)                                          \
      LOAD_B_FRAG(b10_, Barr, ldb, 96, n0_)                                          \
      LOAD_B_FRAG(b11_, Barr, ldb, 96, n1_)                                          \
      WMMA_F16(acc0_, a2_, b00_);                                                    \
      WMMA_F16(acc1_, a2_, b01_);                                                    \
      WMMA_F16(acc0_, a3_, b10_);                                                    \
      WMMA_F16(acc1_, a3_, b11_);                                                    \
      if (HASB) {                                                                    \
        const float bb0_ = (biasptr)[n0_ + (lane & 15)];                             \
        const float bb1_ = (biasptr)[n1_ + (lane & 15)];                             \
        _Pragma("unroll") for (int i_ = 0; i_ < 8; ++i_) {                           \
          acc0_[i_] += bb0_;                                                         \
          acc1_[i_] += bb1_;                                                         \
        }                                                                            \
      }                                                                              \
      STORE_D_FRAG(s.buf, LDC, n0_, acc0_)                                           \
      STORE_D_FRAG(s.buf, LDC, n1_, acc1_)                                           \
    }                                                                                \
  }

// stage a 128x128 f16 weight chunk from global into s.wbufT, TRANSPOSED (N-major)
#define STAGE_WT(srcptr, srcld)                                                      \
  for (int i_ = tid; i_ < 128 * 128; i_ += 256) {                                    \
    const int k_ = i_ >> 7, n_ = i_ & 127;                                           \
    s.wbufT[n_ * WLD + k_] = (srcptr)[k_ * (srcld) + n_];                            \
  }

// ---------------- weight f32 -> f16 prep kernel ----------------
__global__ void nga_cvt_weights(const float* __restrict__ win,  half_t* __restrict__ hwin,  int nwin,
                                const float* __restrict__ wout, half_t* __restrict__ hwout, int nwout) {
  const int i = blockIdx.x * blockDim.x + threadIdx.x;
  const int stride = gridDim.x * blockDim.x;
  for (int j = i; j < nwin;  j += stride) hwin[j]  = (half_t)win[j];
  for (int j = i; j < nwout; j += stride) hwout[j] = (half_t)wout[j];
}

// ---------------- fused NeighborGatedAttention kernel: one block per token ----------------
__global__ __launch_bounds__(256) void nga_fused_kernel(
    const float* __restrict__ gG,    const int* __restrict__ gMask,
    const float* __restrict__ gR,    const float* __restrict__ gSw,
    const half_t* __restrict__ hWin, const float* __restrict__ gBin,
    const half_t* __restrict__ hWout,const float* __restrict__ gBout,
    const float* __restrict__ gLnG,  const float* __restrict__ gLnB,
    float* __restrict__ gOut) {
  __shared__ Smem s;

  const int tid  = threadIdx.x;
  const int lane = tid & 31;
  const int wave = tid >> 5;
  const int b    = blockIdx.x;

  // ---- load token tile (vectorized) + per-neighbor scalars ----
  {
    const float4* G4 = (const float4*)(gG + (size_t)b * (NNEI * EMB));
    for (int i = tid; i < (NNEI * EMB) / 4; i += 256) {
      const float4 v = G4[i];
      const int e = i * 4;
      s.resid[e + 0] = v.x; s.resid[e + 1] = v.y;
      s.resid[e + 2] = v.z; s.resid[e + 3] = v.w;
      const int xb = (e >> 7) * LDH + (e & 127);
      s.xh[xb + 0] = (half_t)v.x; s.xh[xb + 1] = (half_t)v.y;
      s.xh[xb + 2] = (half_t)v.z; s.xh[xb + 3] = (half_t)v.w;
    }
  }
  if (tid < NNEI) {
    s.s_sw[tid]   = gSw[(size_t)b * NNEI + tid];
    s.s_mask[tid] = (gMask[(size_t)b * NNEI + tid] != 0) ? 1.0f : 0.0f;
    const float* rp = gR + ((size_t)b * NNEI + tid) * 3;
    s.s_r[tid * 4 + 0] = rp[0];
    s.s_r[tid * 4 + 1] = rp[1];
    s.s_r[tid * 4 + 2] = rp[2];
  }
  __syncthreads();

  for (int l = 0; l < LNUM; ++l) {
    // ======== QKV projection + per-row l2 normalize ========
    for (int c = 0; c < 3; ++c) {
      STAGE_WT(hWin + (size_t)l * EMB * 3 * HID + c * HID, 3 * HID)
      __syncthreads();
      GEMM128(s.xh, LDH, s.wbufT, WLD, true, gBin + l * 3 * HID + c * HID)
      __syncthreads();
      const float scale_extra = (c == 0) ? SCALING : 1.0f;
      for (int r = wave; r < NNEI; r += 8) {
        float v4[4];
        float ss = 0.f;
#pragma unroll
        for (int qq = 0; qq < 4; ++qq) {
          const float t = s.buf[r * LDC + lane + 32 * qq];
          v4[qq] = t;
          ss += t * t;
        }
        ss = wave_sum(ss);
        const float sc = scale_extra / fmaxf(sqrtf(ss), 1e-12f);
#pragma unroll
        for (int qq = 0; qq < 4; ++qq) {
          const int cc = lane + 32 * qq;
          const half_t hv = (half_t)(v4[qq] * sc);
          if (c == 0)      s.qh[r * LDH + cc] = hv;
          else if (c == 1) s.kh[r * LDH + cc] = hv;
          else             s.vhT[cc * LDH + r] = hv;   // store v transposed for attn@v
        }
      }
      __syncthreads();
    }

    // ======== attn = q @ k^T  (kh row-major == already the B^T layout) ========
    GEMM128(s.qh, LDH, s.kh, LDH, false, gBout)
    __syncthreads();

    // ======== softmax with switch-smoothing, mask, sw_ij * angular gate ========
    for (int r = wave; r < NNEI; r += 8) {
      const float swr = s.s_sw[r];
      const float rx = s.s_r[r * 4 + 0], ry = s.s_r[r * 4 + 1], rz = s.s_r[r * 4 + 2];
      const float rowm = s.s_mask[r];
      float v4[4];
      float mx = -3.4e38f;
#pragma unroll
      for (int qq = 0; qq < 4; ++qq) {
        const int m = lane + 32 * qq;
        const float sij = swr * s.s_sw[m];
        const float t = (s.buf[r * LDC + m] + SHIFT) * sij - SHIFT;
        v4[qq] = t;
        mx = fmaxf(mx, t);
      }
      mx = wave_max(mx);
      float ssum = 0.f;
#pragma unroll
      for (int qq = 0; qq < 4; ++qq) {
        v4[qq] = __expf(v4[qq] - mx);
        ssum += v4[qq];
      }
      ssum = wave_sum(ssum);
      const float inv = rowm / ssum;
#pragma unroll
      for (int qq = 0; qq < 4; ++qq) {
        const int m = lane + 32 * qq;
        const float ang = rx * s.s_r[m * 4 + 0] + ry * s.s_r[m * 4 + 1] + rz * s.s_r[m * 4 + 2];
        const float w = v4[qq] * inv * (swr * s.s_sw[m]) * ang;
        s.qh[r * LDH + m] = (half_t)w;   // attn weights overwrite q (already consumed)
      }
    }
    __syncthreads();

    // ======== o = attn @ v (vhT is v transposed => B^T layout) ========
    GEMM128(s.qh, LDH, s.vhT, LDH, false, gBout)
    __syncthreads();
    for (int i = tid; i < NNEI * HID; i += 256)
      s.xh[(i >> 7) * LDH + (i & 127)] = (half_t)s.buf[i];   // o(f16) overwrites xh
    __syncthreads();

    // ======== out = o @ W_out + b_out ========
    STAGE_WT(hWout + (size_t)l * HID * EMB, EMB)
    __syncthreads();
    GEMM128(s.xh, LDH, s.wbufT, WLD, true, gBout + l * EMB)
    __syncthreads();

    // ======== residual add + layernorm -> next x (f32 + f16 copies) ========
    const float* gv = gLnG + l * EMB;
    const float* bv = gLnB + l * EMB;
    for (int r = wave; r < NNEI; r += 8) {
      float y[4];
      float s1 = 0.f, s2 = 0.f;
#pragma unroll
      for (int qq = 0; qq < 4; ++qq) {
        const int cc = lane + 32 * qq;
        const float t = s.resid[r * EMB + cc] + s.buf[r * LDC + cc];
        y[qq] = t;
        s1 += t;
        s2 += t * t;
      }
      s1 = wave_sum(s1) * (1.0f / EMB);
      s2 = wave_sum(s2) * (1.0f / EMB);
      const float invsd = rsqrtf(s2 - s1 * s1 + LN_EPS);
#pragma unroll
      for (int qq = 0; qq < 4; ++qq) {
        const int cc = lane + 32 * qq;
        const float o = (y[qq] - s1) * invsd * gv[cc] + bv[cc];
        s.resid[r * EMB + cc] = o;
        s.xh[r * LDH + cc] = (half_t)o;
      }
    }
    __syncthreads();
  }

  // ---- write final x (vectorized) ----
  {
    float4* O4 = (float4*)(gOut + (size_t)b * (NNEI * EMB));
    for (int i = tid; i < (NNEI * EMB) / 4; i += 256) {
      const int e = i * 4;
      float4 v;
      v.x = s.resid[e + 0]; v.y = s.resid[e + 1];
      v.z = s.resid[e + 2]; v.w = s.resid[e + 3];
      O4[i] = v;
    }
  }
}

// ---------------- host launcher ----------------
extern "C" void kernel_launch(void* const* d_in, const int* in_sizes, int n_in,
                              void* d_out, int out_size, void* d_ws, size_t ws_size,
                              hipStream_t stream) {
  const float* input_G  = (const float*)d_in[0];
  const int*   nei_mask = (const int*)  d_in[1];
  const float* input_r  = (const float*)d_in[2];
  const float* sw       = (const float*)d_in[3];
  const float* W_in     = (const float*)d_in[4];
  const float* b_in     = (const float*)d_in[5];
  const float* W_out    = (const float*)d_in[6];
  const float* b_out    = (const float*)d_in[7];
  const float* ln_g     = (const float*)d_in[8];
  const float* ln_b     = (const float*)d_in[9];
  float* out = (float*)d_out;

  const int nwin  = LNUM * EMB * 3 * HID;   // 98304
  const int nwout = LNUM * HID * EMB;       // 32768
  half_t* hWin  = (half_t*)d_ws;
  half_t* hWout = hWin + nwin;

  nga_cvt_weights<<<256, 256, 0, stream>>>(W_in, hWin, nwin, W_out, hWout, nwout);

  const int Bn = in_sizes[0] / (NNEI * EMB);
  nga_fused_kernel<<<Bn, 256, 0, stream>>>(
      input_G, nei_mask, input_r, sw, hWin, b_in, hWout, b_out, ln_g, ln_b, out);
}
